// NNConvModel_10333691314295
// MI455X (gfx1250) — matrix operand order repaired and checked by reading, not compiled, not run on hardware
//
#include <hip/hip_runtime.h>
#include <hip/hip_bf16.h>
#include <stdint.h>

#define LEAK 0.1f

typedef __attribute__((ext_vector_type(8)))  float  v8f;
typedef __attribute__((ext_vector_type(16))) __bf16 v16bf;

union FragBF { uint4 u[2]; v16bf v; };

static __device__ __forceinline__ unsigned short f2bf(float f) {
  unsigned int u = __float_as_uint(f);
  unsigned int r = 0x7FFFu + ((u >> 16) & 1u);
  return (unsigned short)((u + r) >> 16);
}

// leaky_relu(v, 0.1) == max(v, 0.1*v) exactly (0 < LEAK < 1):
// lowers to v_mul + v_max_num_f32, avoiding cmp+cndmask chains.
static __device__ __forceinline__ float lrelu(float v) {
  return fmaxf(v, LEAK * v);
}

static __device__ __forceinline__ void load_bfrag(FragBF& F, const unsigned short* base,
                                                  int group, int lane) {
  const uint4* p = (const uint4*)(base + ((size_t)group * 32 + lane) * 16);
  F.u[0] = p[0];
  F.u[1] = p[1];
}

// ---------------- BatchNorm statistics (biased, training mode) ----------------
__global__ __launch_bounds__(256) void bn_stats_kernel(const float* __restrict__ src,
                                                       int rows, int cols,
                                                       float* __restrict__ meanOut,
                                                       float* __restrict__ rsOut) {
  __shared__ float sh[256], sh2[256];
  int c = blockIdx.x;
  float s = 0.f, s2 = 0.f;
  for (int r = threadIdx.x; r < rows; r += 256) {
    float v = src[(size_t)r * cols + c];
    s += v; s2 += v * v;
  }
  sh[threadIdx.x] = s; sh2[threadIdx.x] = s2;
  __syncthreads();
  for (int o = 128; o > 0; o >>= 1) {
    if ((int)threadIdx.x < o) {
      sh[threadIdx.x]  += sh[threadIdx.x + o];
      sh2[threadIdx.x] += sh2[threadIdx.x + o];
    }
    __syncthreads();
  }
  if (threadIdx.x == 0) {
    float inv = 1.f / (float)rows;
    float m = sh[0] * inv;
    float var = sh2[0] * inv - m * m;
    meanOut[c] = m;
    rsOut[c] = rsqrtf(var + 1e-5f);
  }
}

__global__ __launch_bounds__(256) void apply_bn_x_kernel(const float* __restrict__ x,
                                                         const float* __restrict__ m,
                                                         const float* __restrict__ rs,
                                                         const float* __restrict__ g,
                                                         const float* __restrict__ b,
                                                         float* __restrict__ xo, int N) {
  int idx = blockIdx.x * 256 + threadIdx.x;
  if (idx >= N * 32) return;
  int n = idx >> 5, c = idx & 31;
  xo[idx] = (c < 16) ? (x[n * 16 + c] - m[c]) * rs[c] * g[c] + b[c] : 0.f;
}

__global__ __launch_bounds__(256) void apply_bn_e_kernel(const float* __restrict__ e,
                                                         const float* __restrict__ m,
                                                         const float* __restrict__ rs,
                                                         const float* __restrict__ g,
                                                         const float* __restrict__ b,
                                                         float* __restrict__ eo, int E) {
  int idx = blockIdx.x * 256 + threadIdx.x;
  if (idx >= E * 10) return;
  int c = idx % 10;
  eo[idx] = (e[idx] - m[c]) * rs[c] * g[c] + b[c];
}

__global__ __launch_bounds__(256) void zero_kernel(float* __restrict__ p, int n) {
  int i = blockIdx.x * 256 + threadIdx.x;
  if (i < n) p[i] = 0.f;
}

// ---- Re-shuffle a row-major f32 weight [rows, cols] into bf16 WMMA B-fragment order ----
// Layout: [kchunk][ntile(cols/16)][lane(32)][t(16)], K_global = kc*32 + 8*(lane>=16) + (t<8 ? t : 8+t),
// N = tile*16 + (lane&15).  Rows >= `rows` are zero-padded.
__global__ __launch_bounds__(256) void make_frag_kernel(const float* __restrict__ w,
                                                        int rows, int cols, int kchunks,
                                                        unsigned short* __restrict__ out) {
  int ntiles = cols / 16;
  int total = kchunks * ntiles * 512;
  for (int idx = blockIdx.x * 256 + threadIdx.x; idx < total; idx += gridDim.x * 256) {
    int t = idx & 15;
    int lane = (idx >> 4) & 31;
    int tk = idx >> 9;
    int tile = tk % ntiles;
    int kc = tk / ntiles;
    int K = kc * 32 + ((lane >= 16) ? 8 : 0) + ((t < 8) ? t : t + 8);
    int col = tile * 16 + (lane & 15);
    float v = (K < rows) ? w[(size_t)K * cols + col] : 0.f;
    out[idx] = f2bf(v);
  }
}

// ---------------- Fused NNConv edge kernel ----------------
// Per 16-edge tile (one wave): h1 = leaky(e@w1+b1) -> bf16 A-frag; loop i<nin:
// WMMA h1·w2[:, i*32+16h .. +15] -> +b2 -> leaky -> * x[src][i] accumulate;
// finally atomically scatter msg into agg[dst].
__global__ __launch_bounds__(256) void edge_mp_kernel(
    const float* __restrict__ x, const float* __restrict__ ebn,
    const int* __restrict__ srcI, const int* __restrict__ dstI,
    const float* __restrict__ w1, const float* __restrict__ b1,
    const unsigned short* __restrict__ w2f, const float* __restrict__ b2,
    float* agg, int E, int nin) {
  const int tid = threadIdx.x;
  const int lane = tid & 31, w = tid >> 5;
  const int laneHi = lane >> 4;
  const int laneN = lane & 15;
  const int M = laneN;

  __shared__ float b2s[1024];
  __shared__ __align__(16) unsigned short h1s[8][512];  // 16 rows x 32 (K-padded) bf16
  __shared__ __align__(16) float xsT[8][512];           // transposed x_src: [i][row]
  __shared__ int dsts[8][16];

  const int nout = nin * 32;
  for (int i = tid; i < nout; i += 256) b2s[i] = b2[i];

  const int tileBase = (blockIdx.x * 8 + w) * 16;

  if (lane < 16) {
    int e = tileBase + lane; if (e >= E) e = E - 1;
    dsts[w][lane] = dstI[e];
  }
  // x_src transpose fill (32 i-values x 16 rows)
  #pragma unroll 4
  for (int k = 0; k < 16; ++k) {
    int idx = k * 32 + lane;
    int r = idx & 15, i = idx >> 4;
    int e = tileBase + r; if (e >= E) e = E - 1;
    int s = srcI[e];
    xsT[w][i * 16 + r] = x[(size_t)s * 32 + i];
  }
  // h1 = leaky(e@w1+b1), bf16, K-padded to 32
  for (int k = 0; k < 16; ++k) {
    int idx = k * 32 + lane;
    int r = idx >> 5, c = idx & 31;
    float v = 0.f;
    if (c < nin) {
      int e = tileBase + r; if (e >= E) e = E - 1;
      v = b1[c];
      const float* erow = ebn + (size_t)e * 10;
      #pragma unroll
      for (int q = 0; q < 10; ++q) v += erow[q] * w1[q * nin + c];
      v = lrelu(v);
    }
    h1s[w][r * 32 + c] = f2bf(v);
  }
  __syncthreads();

  // A fragment: 16x32 bf16 per ISA layout
  FragBF A;
  {
    const unsigned short* hp = h1s[w] + M * 32 + laneHi * 8;
    A.u[0] = *(const uint4*)(hp);
    A.u[1] = *(const uint4*)(hp + 16);
  }

  float acc[2][8];
  #pragma unroll
  for (int h = 0; h < 2; ++h)
    #pragma unroll
    for (int v = 0; v < 8; ++v) acc[h][v] = 0.f;

  for (int i = 0; i < nin; ++i) {
    const float* xp = xsT[w] + i * 16 + laneHi * 8;
    float4 xa = *(const float4*)xp;
    float4 xb = *(const float4*)(xp + 4);
    float xs[8] = {xa.x, xa.y, xa.z, xa.w, xb.x, xb.y, xb.z, xb.w};
    #pragma unroll
    for (int h = 0; h < 2; ++h) {
      FragBF B;
      load_bfrag(B, w2f, i * 2 + h, lane);
      float bval = b2s[i * 32 + h * 16 + laneN];
      v8f c0 = {};
      v8f X = __builtin_amdgcn_wmma_f32_16x16x32_bf16(false, A.v, false, B.v,
                                                      (short)0, c0, false, false);
      #pragma unroll
      for (int v = 0; v < 8; ++v) {
        float t = lrelu(X[v] + bval);
        acc[h][v] += xs[v] * t;
      }
    }
  }

  // Scatter msg -> agg[dst] (segment_sum via f32 atomics)
  #pragma unroll
  for (int v = 0; v < 8; ++v) {
    int row = v + laneHi * 8;
    int e = tileBase + row;
    if (e < E) {
      float* base = agg + (size_t)dsts[w][row] * 32;
      __hip_atomic_fetch_add(base + laneN, acc[0][v], __ATOMIC_RELAXED, __HIP_MEMORY_SCOPE_AGENT);
      __hip_atomic_fetch_add(base + 16 + laneN, acc[1][v], __ATOMIC_RELAXED, __HIP_MEMORY_SCOPE_AGENT);
    }
  }
}

// x_next = agg + x @ root + bias
__global__ __launch_bounds__(256) void root_kernel(const float* __restrict__ agg,
                                                   const float* __restrict__ xc,
                                                   const float* __restrict__ rootw,
                                                   const float* __restrict__ bias,
                                                   float* __restrict__ xn, int N, int nin) {
  int idx = blockIdx.x * 256 + threadIdx.x;
  if (idx >= N * 32) return;
  int n = idx >> 5, o = idx & 31;
  float s = agg[idx] + bias[o];
  const float* xr = xc + (size_t)n * 32;
  for (int i = 0; i < nin; ++i) s += xr[i] * rootw[i * 32 + o];
  xn[idx] = s;
}

// ---------------- Fused edge-prediction MLP ----------------
// z = [x_src(32) | x_dst(32) | e(10) | pad to 96]; layers 74->64->32 via WMMA,
// 32->16->8->2 scalar per lane.
__global__ __launch_bounds__(256) void pred_kernel(
    const float* __restrict__ x, const float* __restrict__ ebn,
    const int* __restrict__ srcI, const int* __restrict__ dstI,
    const unsigned short* __restrict__ pf0, const float* __restrict__ pb0,
    const unsigned short* __restrict__ pf1, const float* __restrict__ pb1,
    const float* __restrict__ w2g, const float* __restrict__ pb2,
    const float* __restrict__ w3g, const float* __restrict__ pb3,
    const float* __restrict__ w4g, const float* __restrict__ pb4,
    float* __restrict__ out, int E) {
  const int tid = threadIdx.x;
  const int lane = tid & 31, w = tid >> 5;
  const int laneHi = lane >> 4;
  const int laneN = lane & 15;
  const int M = laneN;

  __shared__ __align__(16) unsigned short zbuf[8][1536];  // 16 x 96 bf16 per wave (reused)
  __shared__ float b0s[64], b1s[32];
  __shared__ float w2s[512], w3s[128], w4s[16], b2s2[16], b3s[8], b4s[2];

  for (int i = tid; i < 64; i += 256) b0s[i] = pb0[i];
  for (int i = tid; i < 32; i += 256) b1s[i] = pb1[i];
  for (int i = tid; i < 512; i += 256) w2s[i] = w2g[i];
  for (int i = tid; i < 128; i += 256) w3s[i] = w3g[i];
  if (tid < 16) { w4s[tid] = w4g[tid]; b2s2[tid] = pb2[tid]; }
  if (tid < 8) b3s[tid] = pb3[tid];
  if (tid < 2) b4s[tid] = pb4[tid];

  const int tileBase = (blockIdx.x * 8 + w) * 16;
  unsigned short* z = zbuf[w];

  // fill z: 16 rows x 96 cols (bf16, row stride 96)
  for (int k = 0; k < 48; ++k) {
    int idx = k * 32 + lane;
    int c = idx >> 4, r = idx & 15;
    int e = tileBase + r; if (e >= E) e = E - 1;
    float v;
    if (c < 32)      v = x[(size_t)srcI[e] * 32 + c];
    else if (c < 64) v = x[(size_t)dstI[e] * 32 + (c - 32)];
    else if (c < 74) v = ebn[(size_t)e * 10 + (c - 64)];
    else             v = 0.f;
    z[r * 96 + c] = f2bf(v);
  }
  __syncthreads();

  // A fragments for K-chunks 0..2
  FragBF A0, A1, A2;
  {
    const unsigned short* zp = z + M * 96 + laneHi * 8;
    A0.u[0] = *(const uint4*)(zp +  0); A0.u[1] = *(const uint4*)(zp + 16);
    A1.u[0] = *(const uint4*)(zp + 32); A1.u[1] = *(const uint4*)(zp + 48);
    A2.u[0] = *(const uint4*)(zp + 64); A2.u[1] = *(const uint4*)(zp + 80);
  }

  // Layer 1: 96(K) -> 64(N), write bf16 z2 (row stride 64)
  unsigned short* z2 = z;
  #pragma unroll
  for (int nt = 0; nt < 4; ++nt) {
    FragBF B;
    v8f c = {};
    load_bfrag(B, pf0, 0 * 4 + nt, lane);
    c = __builtin_amdgcn_wmma_f32_16x16x32_bf16(false, A0.v, false, B.v, (short)0, c, false, false);
    load_bfrag(B, pf0, 1 * 4 + nt, lane);
    c = __builtin_amdgcn_wmma_f32_16x16x32_bf16(false, A1.v, false, B.v, (short)0, c, false, false);
    load_bfrag(B, pf0, 2 * 4 + nt, lane);
    c = __builtin_amdgcn_wmma_f32_16x16x32_bf16(false, A2.v, false, B.v, (short)0, c, false, false);
    float bv = b0s[nt * 16 + laneN];
    #pragma unroll
    for (int v = 0; v < 8; ++v) {
      float t = lrelu(c[v] + bv);
      z2[(v + laneHi * 8) * 64 + nt * 16 + laneN] = f2bf(t);
    }
  }
  __syncthreads();

  // Layer 2: 64(K) -> 32(N), write f32 z3 (row stride 32)
  FragBF C0, C1;
  {
    const unsigned short* zp = z2 + M * 64 + laneHi * 8;
    C0.u[0] = *(const uint4*)(zp +  0); C0.u[1] = *(const uint4*)(zp + 16);
    C1.u[0] = *(const uint4*)(zp + 32); C1.u[1] = *(const uint4*)(zp + 48);
  }
  float* z3 = (float*)z;
  #pragma unroll
  for (int nt = 0; nt < 2; ++nt) {
    FragBF B;
    v8f c = {};
    load_bfrag(B, pf1, 0 * 2 + nt, lane);
    c = __builtin_amdgcn_wmma_f32_16x16x32_bf16(false, C0.v, false, B.v, (short)0, c, false, false);
    load_bfrag(B, pf1, 1 * 2 + nt, lane);
    c = __builtin_amdgcn_wmma_f32_16x16x32_bf16(false, C1.v, false, B.v, (short)0, c, false, false);
    float bv = b1s[nt * 16 + laneN];
    #pragma unroll
    for (int v = 0; v < 8; ++v)
      z3[(v + laneHi * 8) * 32 + nt * 16 + laneN] = lrelu(c[v] + bv);
  }
  __syncthreads();

  // Tail: 32 -> 16 -> 8 -> 2 (scalar, lanes 0..15, one edge each)
  if (lane < 16) {
    int e = tileBase + lane;
    if (e < E) {
      const float* zr = z3 + lane * 32;
      float h3[16];
      #pragma unroll
      for (int o = 0; o < 16; ++o) {
        float s = b2s2[o];
        #pragma unroll
        for (int i = 0; i < 32; ++i) s += zr[i] * w2s[i * 16 + o];
        h3[o] = lrelu(s);
      }
      float h4[8];
      #pragma unroll
      for (int o = 0; o < 8; ++o) {
        float s = b3s[o];
        #pragma unroll
        for (int i = 0; i < 16; ++i) s += h3[i] * w3s[i * 8 + o];
        h4[o] = lrelu(s);
      }
      #pragma unroll
      for (int o = 0; o < 2; ++o) {
        float s = b4s[o];
        #pragma unroll
        for (int i = 0; i < 8; ++i) s += h4[i] * w4s[i * 2 + o];
        out[(size_t)e * 2 + o] = s;
      }
    }
  }
}

// ---------------------------------------------------------------------------
extern "C" void kernel_launch(void* const* d_in, const int* in_sizes, int n_in,
                              void* d_out, int out_size, void* d_ws, size_t ws_size,
                              hipStream_t stream) {
  (void)n_in; (void)out_size; (void)ws_size;
  // Two possible flattening conventions: dict insertion order ('x' first, 320000)
  // vs jax tree_flatten sorted keys ('e' first, 1000000). Detect from in_sizes[0].
  const bool sorted = in_sizes[0] > 500000;
  const float *x, *e, *bng, *bnb, *beg, *beb;
  const float *w1[3], *b1[3], *w2[3], *b2[3], *rootw[3], *rbias[3];
  const float *pw[5], *pb[5];
  const int* ei = (const int*)d_in[1];
  auto F = [&](int i) { return (const float*)d_in[i]; };
  if (!sorted) {
    x = F(0); e = F(2);
    bng = F(4); bnb = F(5); beg = F(6); beb = F(7);
    int b = 8;
    for (int l = 0; l < 3; ++l) {
      w1[l] = F(b); b1[l] = F(b + 1); w2[l] = F(b + 2);
      b2[l] = F(b + 3); rootw[l] = F(b + 4); rbias[l] = F(b + 5);
      b += 6;
    }
    for (int p = 0; p < 5; ++p) { pw[p] = F(26 + 2 * p); pb[p] = F(27 + 2 * p); }
  } else {
    e = F(0);
    beg = F(2); beb = F(3); bng = F(4); bnb = F(5);
    int b = 6;
    for (int l = 0; l < 3; ++l) {
      b1[l] = F(b); b2[l] = F(b + 1); rbias[l] = F(b + 2);
      rootw[l] = F(b + 3); w1[l] = F(b + 4); w2[l] = F(b + 5);
      b += 6;
    }
    for (int p = 0; p < 5; ++p) { pw[p] = F(24 + 2 * p); pb[p] = F(25 + 2 * p); }
    x = F(34);
  }
  const int E = in_sizes[1] / 2;
  const int N = in_sizes[sorted ? 34 : 0] / 16;

  char* ws = (char*)d_ws;
  size_t off = 0;
  auto alloc = [&](size_t bytes) -> void* {
    void* p = ws + off;
    off += (bytes + 255) & ~(size_t)255;
    return p;
  };
  float* ebn = (float*)alloc((size_t)E * 10 * 4);
  float* xA  = (float*)alloc((size_t)N * 32 * 4);
  float* xB  = (float*)alloc((size_t)N * 32 * 4);
  float* agg = (float*)alloc((size_t)N * 32 * 4);
  float* mx  = (float*)alloc(64 * 4);
  float* rx  = (float*)alloc(64 * 4);
  float* me  = (float*)alloc(64 * 4);
  float* re  = (float*)alloc(64 * 4);
  unsigned short* w2f[3];
  w2f[0] = (unsigned short*)alloc((size_t)32 * 512 * 2);
  w2f[1] = (unsigned short*)alloc((size_t)64 * 512 * 2);
  w2f[2] = (unsigned short*)alloc((size_t)64 * 512 * 2);
  unsigned short* pf0 = (unsigned short*)alloc((size_t)12 * 512 * 2);
  unsigned short* pf1 = (unsigned short*)alloc((size_t)4 * 512 * 2);

  bn_stats_kernel<<<16, 256, 0, stream>>>(x, N, 16, mx, rx);
  bn_stats_kernel<<<10, 256, 0, stream>>>(e, E, 10, me, re);
  apply_bn_x_kernel<<<(N * 32 + 255) / 256, 256, 0, stream>>>(x, mx, rx, bng, bnb, xA, N);
  apply_bn_e_kernel<<<(E * 10 + 255) / 256, 256, 0, stream>>>(e, me, re, beg, beb, ebn, E);

  make_frag_kernel<<<64, 256, 0, stream>>>(w2[0], 16, 512, 1, w2f[0]);
  make_frag_kernel<<<128, 256, 0, stream>>>(w2[1], 32, 1024, 1, w2f[1]);
  make_frag_kernel<<<128, 256, 0, stream>>>(w2[2], 32, 1024, 1, w2f[2]);
  make_frag_kernel<<<24, 256, 0, stream>>>(pw[0], 74, 64, 3, pf0);
  make_frag_kernel<<<8, 256, 0, stream>>>(pw[1], 64, 32, 2, pf1);

  const int* srcI = ei;
  const int* dstI = ei + E;
  float* xc = xA;
  float* xn = xB;
  const int tiles = (E + 15) / 16;
  const int eblocks = (tiles + 7) / 8;
  for (int l = 0; l < 3; ++l) {
    int nin = (l == 0) ? 16 : 32;
    zero_kernel<<<(N * 32 + 255) / 256, 256, 0, stream>>>(agg, N * 32);
    edge_mp_kernel<<<eblocks, 256, 0, stream>>>(xc, ebn, srcI, dstI, w1[l], b1[l],
                                                w2f[l], b2[l], agg, E, nin);
    root_kernel<<<(N * 32 + 255) / 256, 256, 0, stream>>>(agg, xc, rootw[l], rbias[l],
                                                          xn, N, nin);
    float* t = xc; xc = xn; xn = t;
  }

  pred_kernel<<<eblocks, 256, 0, stream>>>(xc, ebn, srcI, dstI, pf0, pb[0], pf1, pb[1],
                                           pw[2], pb[2], pw[3], pb[3], pw[4], pb[4],
                                           (float*)d_out, E);
}
